// ImplicitTokensAligner_62277025792459
// MI455X (gfx1250) — matrix operand order repaired and checked by reading, not compiled, not run on hardware
//
#include <hip/hip_runtime.h>
#include <hip/hip_bf16.h>
#include <math.h>

// ---------------- problem constants (match reference setup) ----------------
static constexpr int Bn   = 16;     // batch
static constexpr int S    = 2048;   // seq len
static constexpr int H    = 2048;   // hidden
static constexpr int KQ   = 64;     // implicit tokens
static constexpr int NH   = 16;     // heads
static constexpr int DH   = 128;    // head dim
static constexpr int SK   = S + KQ; // 2112 output rows
static constexpr int SMAX = 1024;   // max region length (padded stride)

typedef __attribute__((ext_vector_type(16))) _Float16 v16h;
typedef __attribute__((ext_vector_type(8)))  _Float16 v8h;
typedef __attribute__((ext_vector_type(8)))  float    v8f;

// ---------------- WMMA helpers (gfx1250, wave32) ----------------
__device__ __forceinline__ v8f wmma_f16(v16h a, v16h b, v8f c) {
  return __builtin_amdgcn_wmma_f32_16x16x32_f16(false, a, false, b, (short)0, c,
                                                false, false);
}

// A-fragment (16x32 f16, row-major source with given stride):
// lanes 0-15: K = k0+0..15 ; lanes 16-31: K = k0+16..31 (consecutive halves)
__device__ __forceinline__ v16h load_afrag(const _Float16* base, int stride,
                                           int m_row, int k0, int lane) {
  const _Float16* p = base + (size_t)m_row * stride + k0 + ((lane & 16) ? 16 : 0);
  v8h lo = *(const v8h*)p;
  v8h hi = *(const v8h*)(p + 8);
  v16h r;
#pragma unroll
  for (int i = 0; i < 8; ++i) { r[i] = lo[i]; r[i + 8] = hi[i]; }
  return r;
}

// B-fragment (32x16 f16) from a per-lane row pointer (B[k][n] = row[n][k]):
// lanes 0-15: K = k0+0..15 ; lanes 16-31: K = k0+16..31 (consecutive halves)
__device__ __forceinline__ v16h load_bfrag_f16row(const _Float16* rowp, int k0,
                                                  int lane) {
  const _Float16* p = rowp + k0 + ((lane & 16) ? 16 : 0);
  v8h lo = *(const v8h*)p;
  v8h hi = *(const v8h*)(p + 8);
  v16h r;
#pragma unroll
  for (int i = 0; i < 8; ++i) { r[i] = lo[i]; r[i + 8] = hi[i]; }
  return r;
}

// B-fragment from a per-lane fp32 row (convert on the fly)
__device__ __forceinline__ v16h load_bfrag_f32row(const float* rowp, int k0,
                                                  int lane) {
  const float* p = rowp + k0 + ((lane & 16) ? 16 : 0);
  v16h r;
#pragma unroll
  for (int i = 0; i < 4; ++i) {
    float4 v = ((const float4*)p)[i];
    r[4 * i + 0] = (_Float16)v.x;
    r[4 * i + 1] = (_Float16)v.y;
    r[4 * i + 2] = (_Float16)v.z;
    r[4 * i + 3] = (_Float16)v.w;
  }
  return r;
}

// B-fragment from a strided fp32 column (lane-coalesced across N)
__device__ __forceinline__ v16h load_bfrag_f32col(const float* p, int stride) {
  v16h r;
#pragma unroll
  for (int j = 0; j < 16; ++j) r[j] = (_Float16)p[(size_t)j * stride];
  return r;
}

// C/D tile (16x16 f32 accum): VGPR r, lane l -> M = r + 8*(l>=16), N = l%16
__device__ __forceinline__ void store_ctile_f16(v8f acc, _Float16* base,
                                                int stride, int m0, int n0,
                                                int lane, float mul) {
  int col = n0 + (lane & 15);
  int r0  = m0 + ((lane & 16) ? 8 : 0);
#pragma unroll
  for (int r = 0; r < 8; ++r)
    base[(size_t)(r0 + r) * stride + col] = (_Float16)(acc[r] * mul);
}

// ---------------- K1: find start (3rd <think>) and end (last </think> after)
__global__ void k_markers(const int* __restrict__ ids, const int* think_p,
                          const int* endthink_p, int* __restrict__ se) {
  int b = blockIdx.x;
  const int* row = ids + (size_t)b * S;
  int think = *think_p, endth = *endthink_p;
  __shared__ int cnt[257];
  __shared__ int s_start, s_end;
  int t = threadIdx.x;
  int c = 0;
  for (int i = t * 8; i < t * 8 + 8; ++i) c += (row[i] == think);
  cnt[t + 1] = c;
  if (t == 0) { cnt[0] = 0; s_start = 0; s_end = -1; }
  __syncthreads();
  if (t == 0)
    for (int i = 1; i <= 256; ++i) cnt[i] += cnt[i - 1];
  __syncthreads();
  int run = cnt[t];
  for (int i = t * 8; i < t * 8 + 8; ++i) {
    if (row[i] == think) { run++; if (run == 3) s_start = i; }
  }
  __syncthreads();
  int st = s_start;
  for (int i = t * 8; i < t * 8 + 8; ++i)
    if (row[i] == endth && i > st) atomicMax(&s_end, i);
  __syncthreads();
  if (t == 0) { se[2 * b] = st; se[2 * b + 1] = s_end; }
}

// ---------------- K2a: q = implicit @ Wq^T + bq   [KQ, H] fp32
__global__ void k_proj_q(const float* __restrict__ implicit,
                         const float* __restrict__ W,
                         const float* __restrict__ bias,
                         float* __restrict__ q) {
  int idx = blockIdx.x * 256 + threadIdx.x;   // KQ*H = 131072
  int qi = idx >> 11, h = idx & (H - 1);
  const float* xp = implicit + (size_t)qi * H;
  const float* wp = W + (size_t)h * H;
  float s = bias[h];
  for (int j = 0; j < H; j += 4) {
    float4 x = *(const float4*)(xp + j);
    float4 w = *(const float4*)(wp + j);
    s += x.x * w.x + x.y * w.y + x.z * w.z + x.w * w.w;
  }
  q[idx] = s;
}

// ---------------- K2b: A[(n,q),h] = sum_d q[q, n*DH+d] * Wk[n*DH+d, h]  (f16)
__global__ void k_build_A(const float* __restrict__ q,
                          const float* __restrict__ W,
                          _Float16* __restrict__ A) {
  int idx = blockIdx.x * 256 + threadIdx.x;   // 1024*H = 2M
  int nq = idx >> 11, h = idx & (H - 1);
  int n = nq >> 6, qi = nq & 63;
  const float* qp = q + (size_t)qi * H + n * DH;
  const float* wp = W + ((size_t)(H + n * DH)) * H + h;   // Wk rows [H,2H)
  float s = 0.f;
#pragma unroll 8
  for (int d = 0; d < DH; ++d) s += qp[d] * wp[(size_t)d * H];
  A[(size_t)nq * H + h] = (_Float16)s;
}

// ---------------- K2c: convert Wv and Wout to f16
__global__ void k_cvt_w(const float* __restrict__ Wv,
                        const float* __restrict__ Wo,
                        _Float16* __restrict__ Wvf,
                        _Float16* __restrict__ Wof) {
  size_t n = (size_t)H * H;
  for (size_t i = blockIdx.x * 256ull + threadIdx.x; i < n;
       i += (size_t)gridDim.x * 256ull) {
    Wvf[i] = (_Float16)Wv[i];
    Wof[i] = (_Float16)Wo[i];
  }
}

// ---------------- K3: scores[b][(n,q)][s_local] = scale * A . emb[region]
// grid (32 col-blocks of 32, 4 row-blocks of 256, B); block 256 = 8 waves
// k-loop is software-pipelined (double-buffered fragments).
__global__ void __launch_bounds__(256)
k_scores(const float* __restrict__ emb, const _Float16* __restrict__ A,
         const int* __restrict__ se, _Float16* __restrict__ attn) {
  int b = blockIdx.z;
  int start = se[2 * b], end = se[2 * b + 1];
  int L = end - start - 1;
  if (L > SMAX) L = SMAX;
  if (L < 1) return;
  int colblock = blockIdx.x * 32;
  if (colblock >= L) return;
  int lane = threadIdx.x & 31, wave = threadIdx.x >> 5;
  int mi = wave & 3, ni = wave >> 2;            // 4 M-waves x 2 N-waves
  int m0 = blockIdx.y * 256 + mi * 64;
  int n0 = colblock + ni * 16;
  int srow = start + 1 + n0 + (lane & 15);      // <= start+1024 < 2048
  const float* rowp = emb + ((size_t)b * S + srow) * H;
  int mr = lane & 15;
  v8f acc0 = {}, acc1 = {}, acc2 = {}, acc3 = {};
  // prologue
  v16h bf = load_bfrag_f32row(rowp, 0, lane);
  v16h a0 = load_afrag(A, H, m0 + 0  + mr, 0, lane);
  v16h a1 = load_afrag(A, H, m0 + 16 + mr, 0, lane);
  v16h a2 = load_afrag(A, H, m0 + 32 + mr, 0, lane);
  v16h a3 = load_afrag(A, H, m0 + 48 + mr, 0, lane);
  for (int k0 = 0; k0 < H - 32; k0 += 32) {
    int k1 = k0 + 32;
    v16h bfn = load_bfrag_f32row(rowp, k1, lane);
    v16h a0n = load_afrag(A, H, m0 + 0  + mr, k1, lane);
    v16h a1n = load_afrag(A, H, m0 + 16 + mr, k1, lane);
    v16h a2n = load_afrag(A, H, m0 + 32 + mr, k1, lane);
    v16h a3n = load_afrag(A, H, m0 + 48 + mr, k1, lane);
    acc0 = wmma_f16(a0, bf, acc0);
    acc1 = wmma_f16(a1, bf, acc1);
    acc2 = wmma_f16(a2, bf, acc2);
    acc3 = wmma_f16(a3, bf, acc3);
    bf = bfn; a0 = a0n; a1 = a1n; a2 = a2n; a3 = a3n;
  }
  acc0 = wmma_f16(a0, bf, acc0);
  acc1 = wmma_f16(a1, bf, acc1);
  acc2 = wmma_f16(a2, bf, acc2);
  acc3 = wmma_f16(a3, bf, acc3);

  const float scale = 0.08838834764831845f;     // 1/sqrt(128)
  _Float16* out = attn + (size_t)b * 1024 * SMAX;
  store_ctile_f16(acc0, out, SMAX, m0 + 0,  n0, lane, scale);
  store_ctile_f16(acc1, out, SMAX, m0 + 16, n0, lane, scale);
  store_ctile_f16(acc2, out, SMAX, m0 + 32, n0, lane, scale);
  store_ctile_f16(acc3, out, SMAX, m0 + 48, n0, lane, scale);
}

// ---------------- K4: softmax rows of length L, zero-pad to SMAX
__global__ void k_softmax(_Float16* __restrict__ attn,
                          const int* __restrict__ se) {
  int b = blockIdx.y, row = blockIdx.x;
  int start = se[2 * b], end = se[2 * b + 1];
  int L = end - start - 1;
  if (L > SMAX) L = SMAX;
  _Float16* p = attn + ((size_t)b * 1024 + row) * SMAX;
  int t = threadIdx.x;
  if (L < 1) {
    for (int i = 0; i < 4; ++i) p[t + 256 * i] = (_Float16)0.f;
    return;
  }
  __shared__ float red[256];
  float vals[4];
  float m = -INFINITY;
#pragma unroll
  for (int i = 0; i < 4; ++i) {
    int idx = t + 256 * i;
    float v = (idx < L) ? (float)p[idx] : -INFINITY;
    vals[i] = v;
    m = fmaxf(m, v);
  }
  red[t] = m; __syncthreads();
  for (int off = 128; off > 0; off >>= 1) {
    if (t < off) red[t] = fmaxf(red[t], red[t + off]);
    __syncthreads();
  }
  m = red[0]; __syncthreads();
  float sum = 0.f;
#pragma unroll
  for (int i = 0; i < 4; ++i) {
    int idx = t + 256 * i;
    float e = (idx < L) ? __expf(vals[i] - m) : 0.f;
    vals[i] = e; sum += e;
  }
  red[t] = sum; __syncthreads();
  for (int off = 128; off > 0; off >>= 1) {
    if (t < off) red[t] += red[t + off];
    __syncthreads();
  }
  float inv = 1.0f / red[0];
#pragma unroll
  for (int i = 0; i < 4; ++i) {
    int idx = t + 256 * i;
    p[idx] = (_Float16)((idx < L) ? vals[i] * inv : 0.f);
  }
}

// ---------------- K5: U[b][n][q][h] = sum_s attn[n,q,s] * emb[region s][h]
// grid (H/32, NH, B); block 256 = 4 M-waves (16 rows each) x 2 N-waves
// dynamic-trip k-loop, software-pipelined
__global__ void __launch_bounds__(256)
k_ugemm(const float* __restrict__ emb, const _Float16* __restrict__ attn,
        const int* __restrict__ se, _Float16* __restrict__ U) {
  int b = blockIdx.z, n = blockIdx.y;
  int start = se[2 * b], end = se[2 * b + 1];
  int L = end - start - 1;
  if (L > SMAX) L = SMAX;
  if (L < 1) L = 0;
  int kmax = (L + 31) & ~31;
  int lane = threadIdx.x & 31, wave = threadIdx.x >> 5;
  int mi = wave & 3, ni = wave >> 2;
  int m0 = mi * 16;
  int h0 = blockIdx.x * 32 + ni * 16;
  const _Float16* abase = attn + ((size_t)b * 1024 + n * 64) * SMAX;
  int hcol = h0 + (lane & 15);
  int koff = (lane & 16) ? 16 : 0;
  const float* embb = emb + ((size_t)b * S + (start + 1 + koff)) * H + hcol;
  v8f acc = {};
  if (kmax > 0) {
    v16h af = load_afrag(abase, SMAX, m0 + (lane & 15), 0, lane);
    v16h bf = load_bfrag_f32col(embb, H);
    for (int k0 = 0; k0 < kmax - 32; k0 += 32) {
      int k1 = k0 + 32;
      v16h afn = load_afrag(abase, SMAX, m0 + (lane & 15), k1, lane);
      v16h bfn = load_bfrag_f32col(embb + (size_t)k1 * H, H);
      acc = wmma_f16(af, bf, acc);
      af = afn; bf = bfn;
    }
    acc = wmma_f16(af, bf, acc);
  }
  _Float16* ubase = U + ((size_t)(b * NH + n) * KQ) * H;
  store_ctile_f16(acc, ubase, H, m0, h0, lane, 1.0f);
}

// ---------------- K6a: ctx[b][q][n*DH+d] = U_n . Wv_n^T + bv
// grid (DH/32, NH, B); block 256 = 4 M-waves x 2 N-waves; pipelined
__global__ void __launch_bounds__(256)
k_ctx(const _Float16* __restrict__ U, const _Float16* __restrict__ Wvf,
      const float* __restrict__ bias, _Float16* __restrict__ ctx) {
  int b = blockIdx.z, n = blockIdx.y;
  int lane = threadIdx.x & 31, wave = threadIdx.x >> 5;
  int mi = wave & 3, ni = wave >> 2;
  int m0 = mi * 16;
  int d0 = blockIdx.x * 32 + ni * 16;
  int colg = n * DH + d0 + (lane & 15);         // output channel
  const _Float16* abase = U + ((size_t)(b * NH + n) * KQ) * H;
  const _Float16* wrow = Wvf + (size_t)colg * H;
  v8f acc = {};
  v16h af = load_afrag(abase, H, m0 + (lane & 15), 0, lane);
  v16h bf = load_bfrag_f16row(wrow, 0, lane);
  for (int k0 = 0; k0 < H - 32; k0 += 32) {
    int k1 = k0 + 32;
    v16h afn = load_afrag(abase, H, m0 + (lane & 15), k1, lane);
    v16h bfn = load_bfrag_f16row(wrow, k1, lane);
    acc = wmma_f16(af, bf, acc);
    af = afn; bf = bfn;
  }
  acc = wmma_f16(af, bf, acc);
  float bv = bias[2 * H + colg];
#pragma unroll
  for (int r = 0; r < 8; ++r) acc[r] += bv;
  store_ctile_f16(acc, ctx + (size_t)b * KQ * H, H, m0, n * DH + d0, lane, 1.0f);
}

// ---------------- K6b: aligned = ctx @ Wout^T + bout, written straight into
// out_emb rows [start, start+KQ).  grid (H/32, 1, B); block 256; pipelined
__global__ void __launch_bounds__(256)
k_outproj(const _Float16* __restrict__ ctx, const _Float16* __restrict__ Wof,
          const float* __restrict__ bout, const int* __restrict__ se,
          float* __restrict__ out_emb) {
  int b = blockIdx.z;
  int start = se[2 * b];
  int lane = threadIdx.x & 31, wave = threadIdx.x >> 5;
  int mi = wave & 3, ni = wave >> 2;
  int m0 = mi * 16;
  int i0 = blockIdx.x * 32 + ni * 16;
  int col = i0 + (lane & 15);
  const _Float16* abase = ctx + (size_t)b * KQ * H;
  const _Float16* wrow = Wof + (size_t)col * H;
  v8f acc = {};
  v16h af = load_afrag(abase, H, m0 + (lane & 15), 0, lane);
  v16h bf = load_bfrag_f16row(wrow, 0, lane);
  for (int k0 = 0; k0 < H - 32; k0 += 32) {
    int k1 = k0 + 32;
    v16h afn = load_afrag(abase, H, m0 + (lane & 15), k1, lane);
    v16h bfn = load_bfrag_f16row(wrow, k1, lane);
    acc = wmma_f16(af, bf, acc);
    af = afn; bf = bfn;
  }
  acc = wmma_f16(af, bf, acc);
  float bo = bout[col];
  int r0 = m0 + ((lane & 16) ? 8 : 0);
#pragma unroll
  for (int r = 0; r < 8; ++r) {
    int row_out = start + r0 + r;                // q-row within [0,KQ)
    out_emb[((size_t)b * SK + row_out) * H + col] = acc[r] + bo;
  }
}

// ---------------- K7: scatter remaining rows + mask.  grid (SK, B); block 256
__global__ void k_scatter(const float* __restrict__ emb,
                          const int* __restrict__ amask,
                          const float* __restrict__ pad,
                          const int* __restrict__ se,
                          float* __restrict__ out_emb,
                          int* __restrict__ out_mask) {
  int j = blockIdx.x, b = blockIdx.y;
  int start = se[2 * b], end = se[2 * b + 1];
  int t = threadIdx.x;
  int* mrow = out_mask + (size_t)b * SK + j;
  if (j >= start && j < start + KQ) {           // written by k_outproj
    if (t == 0) *mrow = 1;
    return;
  }
  int src = (j < start) ? j : (j - start - KQ + end + 1);
  bool valid = (src >= 0) && (src < S);
  float* orow = out_emb + ((size_t)b * SK + j) * H;
  if (valid) {
    const float* irow = emb + ((size_t)b * S + src) * H;
#pragma unroll
    for (int u = 0; u < 2; ++u) {
      int off = (t + 256 * u) * 4;
      *(float4*)(orow + off) = *(const float4*)(irow + off);
    }
    if (t == 0) *mrow = amask[(size_t)b * S + src];
  } else {
#pragma unroll
    for (int u = 0; u < 2; ++u) {
      int off = (t + 256 * u) * 4;
      *(float4*)(orow + off) = *(const float4*)(pad + off);
    }
    if (t == 0) *mrow = 0;
  }
}

// ---------------- launcher ----------------
extern "C" void kernel_launch(void* const* d_in, const int* in_sizes, int n_in,
                              void* d_out, int out_size, void* d_ws,
                              size_t ws_size, hipStream_t stream) {
  (void)in_sizes; (void)n_in; (void)out_size; (void)ws_size;
  const float* emb      = (const float*)d_in[0];
  const int*   ids      = (const int*)d_in[1];
  const int*   amask    = (const int*)d_in[2];
  const float* implicit = (const float*)d_in[3];
  const float* W        = (const float*)d_in[4];   // [3H, H]
  const float* bias     = (const float*)d_in[5];   // [3H]
  const float* Wout     = (const float*)d_in[6];   // [H, H]
  const float* bout     = (const float*)d_in[7];   // [H]
  const float* pad      = (const float*)d_in[8];   // [H]
  const int*   thinkp   = (const int*)d_in[9];
  const int*   endthp   = (const int*)d_in[10];

  float* out_emb = (float*)d_out;
  int*   out_mask = (int*)(out_emb + (size_t)Bn * SK * H);

  char* ws = (char*)d_ws;
  int* se = (int*)ws;                      ws += 256;
  float* q = (float*)ws;                   ws += (size_t)KQ * H * 4;
  _Float16* A = (_Float16*)ws;             ws += (size_t)NH * KQ * H * 2;
  _Float16* Wvf = (_Float16*)ws;           ws += (size_t)H * H * 2;
  _Float16* Wof = (_Float16*)ws;           ws += (size_t)H * H * 2;
  _Float16* attn = (_Float16*)ws;          ws += (size_t)Bn * 1024 * SMAX * 2;
  _Float16* U = (_Float16*)ws;             ws += (size_t)Bn * NH * KQ * H * 2;
  _Float16* ctx = (_Float16*)ws;           ws += (size_t)Bn * KQ * H * 2;

  k_markers<<<dim3(Bn), 256, 0, stream>>>(ids, thinkp, endthp, se);
  k_proj_q<<<dim3((KQ * H) / 256), 256, 0, stream>>>(implicit, W, bias, q);
  k_build_A<<<dim3((NH * KQ * H) / 256), 256, 0, stream>>>(q, W, A);
  k_cvt_w<<<dim3(4096), 256, 0, stream>>>(W + (size_t)2 * H * H, Wout, Wvf, Wof);
  k_scores<<<dim3(32, 4, Bn), 256, 0, stream>>>(emb, A, se, attn);
  k_softmax<<<dim3(NH * KQ, Bn), 256, 0, stream>>>(attn, se);
  k_ugemm<<<dim3(H / 32, NH, Bn), 256, 0, stream>>>(emb, attn, se, U);
  k_ctx<<<dim3(DH / 32, NH, Bn), 256, 0, stream>>>(U, Wvf, bias, ctx);
  k_outproj<<<dim3(H / 32, 1, Bn), 256, 0, stream>>>(ctx, Wof, bout, se, out_emb);
  k_scatter<<<dim3(SK, Bn), 256, 0, stream>>>(emb, amask, pad, se, out_emb,
                                              out_mask);
}
